// GRU_33234456936600
// MI455X (gfx1250) — compile-verified, matrix operand order checked by reading, not run on hardware
//
#include <hip/hip_runtime.h>
#include <hip/hip_bf16.h>
#include <stdint.h>

// Problem constants (match reference)
#define B_   64
#define T_   512
#define IN_  512
#define H_   1024
#define G_   (3 * H_)          // 3072
#define BT_  (B_ * T_)         // 32768
#define NBLK_SCAN (H_ / 64)    // 16 persistent blocks for the recurrence

typedef __bf16 bf16_t;
typedef __attribute__((ext_vector_type(16))) __bf16 v16bf;
typedef __attribute__((ext_vector_type(8)))  float  v8f;
typedef __attribute__((__vector_size__(16))) int    v4i_t;

#if __has_builtin(__builtin_amdgcn_global_load_async_to_lds_b128)
#define HAVE_ASYNC_LDS 1
typedef __attribute__((address_space(1))) v4i_t gv4i;   // global-side 16B chunk
typedef __attribute__((address_space(3))) v4i_t lv4i;   // LDS-side 16B chunk
#endif

static __device__ inline void wait_async_le(int n) {
#if __has_builtin(__builtin_amdgcn_s_wait_asynccnt)
    if (n == 0) __builtin_amdgcn_s_wait_asynccnt(0);
    else        __builtin_amdgcn_s_wait_asynccnt(4);
#else
    if (n == 0) asm volatile("s_wait_asynccnt 0" ::: "memory");
    else        asm volatile("s_wait_asynccnt 4" ::: "memory");
#endif
}

static __device__ inline v8f zero8() {
    v8f z = {0.f, 0.f, 0.f, 0.f, 0.f, 0.f, 0.f, 0.f};
    return z;
}

// Build a 16x(32) bf16 A/B fragment for one lane from two 16B chunks.
// Per CDNA5 ISA layout: lanes 0-15 hold K=[0..7,16..23], lanes 16-31 K=[8..15,24..31],
// i.e. chunk0 at elem offset (lane>>4)*8, chunk1 at +16 elems.
static __device__ inline v16bf frag_from(const bf16_t* p0, const bf16_t* p1) {
    union { uint4 u[2]; v16bf v; } x;
    x.u[0] = *reinterpret_cast<const uint4*>(p0);
    x.u[1] = *reinterpret_cast<const uint4*>(p1);
    return x.v;
}

// ---------------------------------------------------------------------------
// Elementwise f32 -> bf16 conversion
// ---------------------------------------------------------------------------
__global__ void cvt_f32_bf16(const float* __restrict__ in, bf16_t* __restrict__ out, int n) {
    int i = blockIdx.x * blockDim.x + threadIdx.x;
    int stride = gridDim.x * blockDim.x;
    for (; i < n; i += stride) out[i] = (bf16_t)in[i];
}

// ---------------------------------------------------------------------------
// Init: zero barrier counters, seed h state (f32 + bf16 mirrors) for both layers
// ---------------------------------------------------------------------------
__global__ void init_state(const float* __restrict__ h0,
                           float* __restrict__ hf_l0, bf16_t* __restrict__ hb_l0,
                           float* __restrict__ hf_l1, bf16_t* __restrict__ hb_l1,
                           int* __restrict__ bar) {
    int i = blockIdx.x * blockDim.x + threadIdx.x;
    if (i < 2) bar[i] = 0;
    const int n = B_ * H_;
    for (int k = i; k < n; k += gridDim.x * blockDim.x) {
        float a = h0[k];     hf_l0[k] = a; hb_l0[k] = (bf16_t)a;
        float b = h0[n + k]; hf_l1[k] = b; hb_l1[k] = (bf16_t)b;
    }
}

// ---------------------------------------------------------------------------
// gi = A[M x K](bf16) * W[G_ x K](bf16)^T + bias  -> out[M x G_] (f32)
// Workgroup: 256 threads = 8 waves, tile 128(M) x 128(N).
// Wave (w>>2) picks M-half (0/64), (w&3) picks N-quarter (0/32/64/96).
// K staged in 32-wide chunks through LDS, double-buffered; on CDNA5 the
// staging uses GLOBAL_LOAD_ASYNC_TO_LDS_B128 tracked by ASYNCcnt.
// ---------------------------------------------------------------------------
template <int K>
__global__ __launch_bounds__(256) void gemm_gi(const bf16_t* __restrict__ A,
                                               const bf16_t* __restrict__ W,
                                               const float*  __restrict__ bias,
                                               float* __restrict__ out) {
    __shared__ __align__(16) bf16_t As[2][128 * 32];
    __shared__ __align__(16) bf16_t Ws[2][128 * 32];

    const int tid  = threadIdx.x;
    const int w    = tid >> 5;
    const int lane = tid & 31;
    const int m_blk = blockIdx.x * 128;
    const int n_blk = blockIdx.y * 128;
    const int wm = (w >> 2) * 64;
    const int wn = (w & 3) * 32;

    v8f acc[4][2];
#pragma unroll
    for (int mt = 0; mt < 4; mt++)
#pragma unroll
        for (int nt = 0; nt < 2; nt++) acc[mt][nt] = zero8();

    const int fr = lane & 15;
    const int fc = (lane >> 4) * 8;

    // Stage one 128x32 A-tile + 128x32 W-tile into LDS buffer `buf`.
    auto stage = [&](int buf, int k0) {
#pragma unroll
        for (int i = 0; i < 2; i++) {
            int idx = tid * 2 + i;        // 0..511 : 512 16B chunks per matrix
            int row = idx >> 2;
            int ch  = (idx & 3) * 8;      // bf16 element offset within row
#ifdef HAVE_ASYNC_LDS
            __builtin_amdgcn_global_load_async_to_lds_b128(
                (gv4i*)&A[(size_t)(m_blk + row) * K + k0 + ch],
                (lv4i*)&As[buf][row * 32 + ch], 0, 0);
            __builtin_amdgcn_global_load_async_to_lds_b128(
                (gv4i*)&W[(size_t)(n_blk + row) * K + k0 + ch],
                (lv4i*)&Ws[buf][row * 32 + ch], 0, 0);
#else
            *reinterpret_cast<uint4*>(&As[buf][row * 32 + ch]) =
                *reinterpret_cast<const uint4*>(&A[(size_t)(m_blk + row) * K + k0 + ch]);
            *reinterpret_cast<uint4*>(&Ws[buf][row * 32 + ch]) =
                *reinterpret_cast<const uint4*>(&W[(size_t)(n_blk + row) * K + k0 + ch]);
#endif
        }
    };

    stage(0, 0);
    constexpr int NC = K / 32;
    for (int c = 0; c < NC; c++) {
        const int k0  = c * 32;
        const int buf = c & 1;
        const bool more = (c + 1) < NC;
        if (more) stage(buf ^ 1, k0 + 32);   // prefetch next chunk into other buffer
#ifdef HAVE_ASYNC_LDS
        // 4 async ops per stage per wave, in-order completion:
        // <=4 outstanding means chunk c is fully in LDS.
        wait_async_le(more ? 4 : 0);
#else
        if (more) {
            __builtin_prefetch(&A[(size_t)(m_blk + (tid >> 1)) * K + k0 + 32], 0, 0);
            __builtin_prefetch(&W[(size_t)(n_blk + (tid >> 1)) * K + k0 + 32], 0, 0);
        }
#endif
        __syncthreads();

        v16bf bfr[2];
#pragma unroll
        for (int nt = 0; nt < 2; nt++) {
            const bf16_t* bp = &Ws[buf][(wn + nt * 16 + fr) * 32 + fc];
            bfr[nt] = frag_from(bp, bp + 16);
        }
#pragma unroll
        for (int mt = 0; mt < 4; mt++) {
            const bf16_t* ap = &As[buf][(wm + mt * 16 + fr) * 32 + fc];
            v16bf af = frag_from(ap, ap + 16);
#pragma unroll
            for (int nt = 0; nt < 2; nt++) {
                acc[mt][nt] = __builtin_amdgcn_wmma_f32_16x16x32_bf16(
                    false, af, false, bfr[nt], (short)0, acc[mt][nt], false, false);
            }
        }
        __syncthreads();
    }

    // Epilogue: C/D layout row = r + (lane>>4)*8, col = lane&15
    const int cr = (lane >> 4) * 8;
    const int cc = lane & 15;
#pragma unroll
    for (int mt = 0; mt < 4; mt++)
#pragma unroll
        for (int nt = 0; nt < 2; nt++)
#pragma unroll
            for (int r = 0; r < 8; r++) {
                int m = m_blk + wm + mt * 16 + r + cr;
                int n = n_blk + wn + nt * 16 + cc;
                out[(size_t)m * G_ + n] = acc[mt][nt][r] + bias[n];
            }
}

// ---------------------------------------------------------------------------
// Persistent GRU recurrence for one layer.
// 16 blocks x 192 threads (6 waves). Block owns h-columns [c0, c0+64).
// Per step: gh strip [64 x 192] = h_prev(bf16) * W_hh rows^T via WMMA,
// gh exchanged through LDS, gate epilogue in f32, fenced global barrier.
// h ping-pong buffers are addressed off single base pointers so the state
// loads lower as global_load (not flat).
// ---------------------------------------------------------------------------
__global__ __launch_bounds__(192) void gru_scan(
    const float*  __restrict__ gi,     // [BT_ x G_] precomputed input gates (f32)
    const bf16_t* __restrict__ Whh,    // [G_ x H_] bf16
    const float*  __restrict__ bhh,    // [G_]
    float*  __restrict__ hf,           // [2][B_ x H_] f32 ping-pong
    bf16_t* __restrict__ hbb,          // [2][B_ x H_] bf16 ping-pong
    bf16_t* __restrict__ out_bf,       // [BT_ x H_] or null (layer-0 output, feeds gi1 GEMM)
    float*  __restrict__ out_f,        // [BT_ x H_] or null (final output)
    float*  __restrict__ hn,           // [B_ x H_] final hidden state slot
    int* __restrict__ bar) {
    __shared__ float gh[64 * 192];

    const int tid  = threadIdx.x;
    const int w    = tid >> 5;        // 0..5
    const int lane = tid & 31;
    const int c0   = blockIdx.x * 64; // owned h-column base
    const int gate = w >> 1;          // 0=r, 1=z, 2=n
    const int wn   = (w & 1) * 32;    // 32-col slice within the gate
    const int fr   = lane & 15;
    const int fc   = (lane >> 4) * 8;

#pragma unroll 1
    for (int t = 0; t < T_; t++) {
        const int cur = t & 1, nxt = cur ^ 1;
        const bf16_t* hprev_b = hbb + (size_t)cur * (B_ * H_);
        const float*  hprev_f = hf  + (size_t)cur * (B_ * H_);
        float*        hnext_f = hf  + (size_t)nxt * (B_ * H_);
        bf16_t*       hnext_b = hbb + (size_t)nxt * (B_ * H_);

        v8f acc[4][2];
#pragma unroll
        for (int mt = 0; mt < 4; mt++)
#pragma unroll
            for (int nt = 0; nt < 2; nt++) acc[mt][nt] = zero8();

#pragma unroll 4
        for (int k0 = 0; k0 < H_; k0 += 32) {
            v16bf bfr[2];
#pragma unroll
            for (int nt = 0; nt < 2; nt++) {
                const bf16_t* bp =
                    &Whh[(size_t)(gate * H_ + c0 + wn + nt * 16 + fr) * H_ + k0 + fc];
                bfr[nt] = frag_from(bp, bp + 16);
            }
#pragma unroll
            for (int mt = 0; mt < 4; mt++) {
                const bf16_t* ap = &hprev_b[(size_t)(mt * 16 + fr) * H_ + k0 + fc];
                v16bf af = frag_from(ap, ap + 16);
#pragma unroll
                for (int nt = 0; nt < 2; nt++) {
                    acc[mt][nt] = __builtin_amdgcn_wmma_f32_16x16x32_bf16(
                        false, af, false, bfr[nt], (short)0, acc[mt][nt], false, false);
                }
            }
        }

        // Dump gh strip to LDS for the cross-gate elementwise stage.
        const int cr = (lane >> 4) * 8;
        const int cc = lane & 15;
#pragma unroll
        for (int mt = 0; mt < 4; mt++)
#pragma unroll
            for (int nt = 0; nt < 2; nt++)
#pragma unroll
                for (int r = 0; r < 8; r++)
                    gh[(mt * 16 + r + cr) * 192 + w * 32 + nt * 16 + cc] = acc[mt][nt][r];
        __syncthreads();

        // Gate epilogue over this block's 64 batches x 64 h-columns.
        for (int idx = tid; idx < 64 * 64; idx += 192) {
            int b = idx >> 6, j = idx & 63;
            int col = c0 + j;
            size_t girow = ((size_t)b * T_ + t) * G_;
            float i_r = gi[girow + col];
            float i_z = gi[girow + H_ + col];
            float i_n = gi[girow + 2 * H_ + col];
            float h_r = gh[b * 192 + j]       + bhh[col];
            float h_z = gh[b * 192 + 64 + j]  + bhh[H_ + col];
            float h_n = gh[b * 192 + 128 + j] + bhh[2 * H_ + col];
            float r = 1.f / (1.f + __expf(-(i_r + h_r)));
            float z = 1.f / (1.f + __expf(-(i_z + h_z)));
            float n = tanhf(i_n + r * h_n);
            float hp = hprev_f[b * H_ + col];
            float hnew = (1.f - z) * hp + z * n;
            hnext_f[b * H_ + col] = hnew;
            hnext_b[b * H_ + col] = (bf16_t)hnew;
            size_t orow = ((size_t)b * T_ + t) * H_ + col;
            if (out_bf) out_bf[orow] = (bf16_t)hnew;
            if (out_f)  out_f[orow]  = hnew;
            if (t == T_ - 1) hn[b * H_ + col] = hnew;
        }

        // Fenced grid barrier across the 16 co-resident blocks.
        __syncthreads();
        if (tid == 0) {
            __threadfence();
            __hip_atomic_fetch_add(bar, 1, __ATOMIC_ACQ_REL, __HIP_MEMORY_SCOPE_AGENT);
            const int target = NBLK_SCAN * (t + 1);
            while (__hip_atomic_load(bar, __ATOMIC_ACQUIRE, __HIP_MEMORY_SCOPE_AGENT) < target)
                __builtin_amdgcn_s_sleep(1);
        }
        __syncthreads();
    }
}

// ---------------------------------------------------------------------------
// Host-side launcher
// ---------------------------------------------------------------------------
extern "C" void kernel_launch(void* const* d_in, const int* in_sizes, int n_in,
                              void* d_out, int out_size, void* d_ws, size_t ws_size,
                              hipStream_t stream) {
    (void)in_sizes; (void)n_in; (void)out_size; (void)ws_size;
    const float* x    = (const float*)d_in[0];
    const float* h0   = (const float*)d_in[1];
    const float* Wih0 = (const float*)d_in[2];
    const float* Whh0 = (const float*)d_in[3];
    const float* bih0 = (const float*)d_in[4];
    const float* bhh0 = (const float*)d_in[5];
    const float* Wih1 = (const float*)d_in[6];
    const float* Whh1 = (const float*)d_in[7];
    const float* bih1 = (const float*)d_in[8];
    const float* bhh1 = (const float*)d_in[9];
    float* out = (float*)d_out;

    uint8_t* ws = (uint8_t*)d_ws;
    size_t off = 0;
    auto carve = [&](size_t bytes) -> void* {
        off = (off + 255) & ~(size_t)255;
        void* p = ws + off;
        off += bytes;
        return p;
    };
    float*  gi      = (float*) carve((size_t)BT_ * G_ * 4);   // reused by both layers
    bf16_t* x_bf    = (bf16_t*)carve((size_t)BT_ * IN_ * 2);
    bf16_t* o0_bf   = (bf16_t*)carve((size_t)BT_ * H_ * 2);
    bf16_t* wih0_bf = (bf16_t*)carve((size_t)G_ * IN_ * 2);
    bf16_t* whh0_bf = (bf16_t*)carve((size_t)G_ * H_ * 2);
    bf16_t* wih1_bf = (bf16_t*)carve((size_t)G_ * H_ * 2);
    bf16_t* whh1_bf = (bf16_t*)carve((size_t)G_ * H_ * 2);
    float*  hf_l0   = (float*) carve((size_t)2 * B_ * H_ * 4);
    float*  hf_l1   = (float*) carve((size_t)2 * B_ * H_ * 4);
    bf16_t* hb_l0   = (bf16_t*)carve((size_t)2 * B_ * H_ * 2);
    bf16_t* hb_l1   = (bf16_t*)carve((size_t)2 * B_ * H_ * 2);
    int*    bar     = (int*)   carve(2 * sizeof(int));

    init_state<<<64, 256, 0, stream>>>(h0, hf_l0, hb_l0, hf_l1, hb_l1, bar);
    cvt_f32_bf16<<<1024, 256, 0, stream>>>(x,    x_bf,    BT_ * IN_);
    cvt_f32_bf16<<<512,  256, 0, stream>>>(Wih0, wih0_bf, G_ * IN_);
    cvt_f32_bf16<<<512,  256, 0, stream>>>(Whh0, whh0_bf, G_ * H_);
    cvt_f32_bf16<<<512,  256, 0, stream>>>(Wih1, wih1_bf, G_ * H_);
    cvt_f32_bf16<<<512,  256, 0, stream>>>(Whh1, whh1_bf, G_ * H_);

    dim3 gdim(BT_ / 128, G_ / 128);

    // Layer 0
    gemm_gi<IN_><<<gdim, 256, 0, stream>>>(x_bf, wih0_bf, bih0, gi);
    gru_scan<<<NBLK_SCAN, 192, 0, stream>>>(gi, whh0_bf, bhh0,
        hf_l0, hb_l0, o0_bf, nullptr, out + (size_t)BT_ * H_, bar);

    // Layer 1
    gemm_gi<H_><<<gdim, 256, 0, stream>>>(o0_bf, wih1_bf, bih1, gi);
    gru_scan<<<NBLK_SCAN, 192, 0, stream>>>(gi, whh1_bf, bhh1,
        hf_l1, hb_l1, nullptr, out,
        out + (size_t)BT_ * H_ + (size_t)B_ * H_, bar + 1);
}
